// AllCostVolume_42219528520114
// MI455X (gfx1250) — matrix operand order repaired and checked by reading, not compiled, not run on hardware
//
#include <hip/hip_runtime.h>

typedef unsigned short u16;
typedef unsigned int   u32;

typedef __attribute__((ext_vector_type(16))) __bf16 bf16x16;
typedef __attribute__((ext_vector_type(8)))  float  f32x8;

#define DEVINL __device__ __forceinline__

static constexpr float BNINV = 0.9999950000374997f; // 1/sqrt(1+1e-5)

// Padded K-strides (bank-conflict avoidance: per-lane dword stride becomes odd)
static constexpr int ST64 = 66, ST128 = 130, ST192 = 194;

// ---------------- workspace layout (bytes) ----------------
static constexpr size_t O_WXYZ = 0;                    // 1024*3*4
static constexpr size_t O_PE   = O_WXYZ + 12288;       // 1024*64*4
static constexpr size_t O_QE   = O_PE   + 262144;
static constexpr size_t O_Q1   = O_QE   + 262144;      // pi_enc per-m part
static constexpr size_t O_COS  = O_Q1   + 262144;      // 1024*1024*4
static constexpr size_t O_RMAX = O_COS  + 4194304;     // 1024*4
static constexpr size_t O_CMAX = O_RMAX + 4096;
static constexpr size_t O_PIF  = O_CMAX + 4096;        // pi_feat 1024*64*4
static constexpr size_t O_KNN  = O_PIF  + 262144;      // 1024*16 int
static constexpr size_t O_PNBF = O_KNN  + 65536;       // 1024*64 bf16
static constexpr size_t O_QNBF = O_PNBF + 131072;
// packed transposed weights, padded row stride (kused+2)
static constexpr size_t O_WT1  = O_QNBF + 131072;      // 128 x ST64  = 16896 B
static constexpr size_t O_WT2  = O_WT1  + 16896;       // 64  x ST128 = 16640 B
static constexpr size_t O_WT3  = O_WT2  + 16640;       // 64  x ST64  =  8448 B
static constexpr size_t O_MT1  = O_WT3  + 8448;        // 128 x ST128 = 33280 B
static constexpr size_t O_MT2  = O_MT1  + 33280;       // 64  x ST128 = 16640 B
static constexpr size_t O_BT1  = O_MT2  + 16640;       // 128 x ST192 = 49664 B
static constexpr size_t O_BT2  = O_BT1  + 49664;       // 64  x ST128 = 16640 B
static constexpr size_t O_VEC  = O_BT2  + 16640;       // 2624 floats

static constexpr int MAIN_WCOPY_DW = (16896+16640+8448+33280+16640)/4; // 22976
static constexpr int S2_WCOPY_DW   = (49664+16640)/4;                  // 16576

// VEC region float offsets
static constexpr int V_WA = 0, V_WB = 128, V_WC = 256, V_WX0 = 384, V_WX1 = 512,
                     V_WX2 = 640, V_WR = 768, V_WCM = 896, V_C1 = 1024,
                     V_C2 = 1152, V_C3 = 1216, V_CM1 = 1280, V_CM2 = 1408,
                     V_CB1 = 1472, V_CB2 = 1600, V_WPIX = 1664, V_CPI = 1856,
                     V_WPC = 1920, V_CPC = 2560;

// main kernel LDS: halves region then float area
static constexpr int LH_WT1 = 0;                 // 128*ST64  = 8448 halves
static constexpr int LH_WT2 = LH_WT1 + 8448;     // 64*ST128  = 8320
static constexpr int LH_WT3 = LH_WT2 + 8320;     // 64*ST64   = 4224
static constexpr int LH_MT1 = LH_WT3 + 4224;     // 128*ST128 = 16640
static constexpr int LH_MT2 = LH_MT1 + 16640;    // 64*ST128  = 8320
static constexpr int LH_STG = LH_MT2 + 8320;     // 8 waves * 16*ST128(=2080) halves
static constexpr int LH_END = LH_STG + 8*2080;   // 62592 halves
static constexpr int L_FBASE = LH_END * 2;       // 125184 bytes

static constexpr int FA_WA=0, FA_WB=128, FA_WC=256, FA_WR=384, FA_WCM=512,
                     FA_C1=640, FA_X1N=768, FA_C2=896, FA_C3=960, FA_CM1=1024,
                     FA_CM2=1152, FA_PEN=1216, FA_P1N=1280, FA_SF=1344,
                     FA_MRG=1984;
static constexpr size_t MAIN_LDS = L_FBASE + (FA_MRG + 16*192) * 4;   // 145408

// stage2 LDS
static constexpr int S2H_BT1 = 0;                  // 128*ST192 = 24832 halves
static constexpr int S2H_BT2 = S2H_BT1 + 24832;    // 64*ST128  = 8320
static constexpr int S2H_STG = S2H_BT2 + 8320;     // 8 waves * 16*ST192(=3104)
static constexpr int S2H_END = S2H_STG + 8*3104;   // 57984 halves
static constexpr int S2_FBASE = S2H_END * 2;       // 115968 bytes
static constexpr size_t S2_LDS = S2_FBASE + (896 + 8*176) * 4;        // 125184

struct InPtrs {
  const float *warped_xyz, *warped_points, *f2_xyz, *f2_points, *lidar_z;
  const float *W1,*b1,*g1,*be1,  *W2,*b2,*g2,*be2,  *W3,*b3,*g3,*be3;   // mlp1
  const float *M1,*mb1,*mg1,*mbe1, *M2,*mb2,*mg2,*mbe2;                  // mlp2
  const float *B1,*bb1,*bg1,*bbe1, *B2,*bb2,*bg2,*bbe2;                  // mlp2b
  const float *Wpi,*pib,*pig,*pibe;                                      // pi_enc
  const float *Wpc,*pcb,*pcg,*pcbe;                                      // pc_enc
};

DEVINL u16 f2bf(float f) {
  union { float f; u32 u; } c; c.f = f;
  u32 u = c.u + 0x7FFFu + ((c.u >> 16) & 1u);
  return (u16)(u >> 16);
}

union ABu { u32 u[8]; bf16x16 v; };

DEVINL void wave_fence() {
  __asm volatile("" ::: "memory");
  __builtin_amdgcn_wave_barrier();
  __asm volatile("" ::: "memory");
}

// A-fragment: 16x32 bf16 tile from row-major [16][ldk] halves (ldk even).
// lane = row%16; lanes 16..31 take K+8; VGPR v: k = (v&3)*2 + (v>>2)*16.
DEVINL bf16x16 load_a_frag(const u16* base, int ldk, int k0) {
  int lane = threadIdx.x & 31;
  int row  = lane & 15;
  int kh   = (lane >> 4) << 3;
  ABu t;
#pragma unroll
  for (int v = 0; v < 8; ++v) {
    int k = k0 + ((v & 3) << 1) + ((v >> 2) << 4) + kh;
    t.u[v] = *reinterpret_cast<const u32*>(base + row * ldk + k);
  }
  return t.v;
}

// B-fragment: 32x16 bf16 from transposed weights WT[cout][ldk] (K contiguous).
// lane = column; lanes 16..31 take K+8; VGPR v: k = (v&3)*2 + (v>>2)*16.
DEVINL bf16x16 load_b_frag(const u16* wt, int ldk, int j0, int k0) {
  int lane = threadIdx.x & 31;
  int col  = j0 + (lane & 15);
  int kh   = (lane >> 4) << 3;
  ABu t;
#pragma unroll
  for (int v = 0; v < 8; ++v) {
    int k = k0 + kh + ((v & 3) << 1) + ((v >> 2) << 4);
    t.u[v] = *reinterpret_cast<const u32*>(wt + col * ldk + k);
  }
  return t.v;
}

DEVINL f32x8 wmma_bf16(bf16x16 a, bf16x16 b, f32x8 c) {
  return __builtin_amdgcn_wmma_f32_16x16x32_bf16(false, a, false, b, (short)0, c,
                                                 false, false);
}

// ---------------------------------------------------------------------------
// Weight prep: transposed bf16 weights (BN-scale folded, padded stride).
// ---------------------------------------------------------------------------
DEVINL void packW(u16* dst, const float* W, const float* g, int cout, int r0,
                  int kused, int idx) {
  int j = idx / kused, k = idx - j * kused;
  dst[j * (kused + 2) + k] = f2bf(W[(r0 + k) * cout + j] * g[j] * BNINV);
}

__global__ void prep_weights(InPtrs P, char* ws) {
  int t0 = blockIdx.x * blockDim.x + threadIdx.x;
  int NT = gridDim.x * blockDim.x;
  u16* wt1 = (u16*)(ws + O_WT1); u16* wt2 = (u16*)(ws + O_WT2);
  u16* wt3 = (u16*)(ws + O_WT3); u16* mt1 = (u16*)(ws + O_MT1);
  u16* mt2 = (u16*)(ws + O_MT2); u16* bt1 = (u16*)(ws + O_BT1);
  u16* bt2 = (u16*)(ws + O_BT2);
  for (int idx = t0; idx < 77824; idx += NT) {
    int i = idx;
    if (i < 8192)                  packW(wt1, P.W1, P.g1, 128, 6,  64, i);
    else if ((i -= 8192)  < 8192)  packW(wt2, P.W2, P.g2,  64, 0, 128, i);
    else if ((i -= 8192)  < 4096)  packW(wt3, P.W3, P.g3,  64, 0,  64, i);
    else if ((i -= 4096)  < 16384) packW(mt1, P.M1, P.mg1,128, 0, 128, i);
    else if ((i -= 16384) < 8192)  packW(mt2, P.M2, P.mg2, 64, 0, 128, i);
    else if ((i -= 8192)  < 24576) packW(bt1, P.B1, P.bg1,128, 0, 192, i);
    else                           packW(bt2, P.B2, P.bg2, 64, 0, 128, i - 24576);
  }
  float* V = (float*)(ws + O_VEC);
  if (t0 < 128) {
    int j = t0;
    float gs1 = P.g1[j] * BNINV;
    V[V_WX0+j] = P.W1[0*128+j]*gs1;  V[V_WX1+j] = P.W1[1*128+j]*gs1;
    V[V_WX2+j] = P.W1[2*128+j]*gs1;  V[V_WA +j] = P.W1[3*128+j]*gs1;
    V[V_WB +j] = P.W1[4*128+j]*gs1;  V[V_WC +j] = P.W1[5*128+j]*gs1;
    V[V_WR +j] = P.W1[70*128+j]*gs1; V[V_WCM+j] = P.W1[71*128+j]*gs1;
    V[V_C1 +j] = P.b1[j]*gs1 + P.be1[j];
    V[V_CM1+j] = P.mb1[j]*P.mg1[j]*BNINV + P.mbe1[j];
    V[V_CB1+j] = P.bb1[j]*P.bg1[j]*BNINV + P.bbe1[j];
  }
  if (t0 < 64) {
    int j = t0;
    V[V_C2 +j] = P.b2[j]*P.g2[j]*BNINV + P.be2[j];
    V[V_C3 +j] = P.b3[j]*P.g3[j]*BNINV + P.be3[j];
    V[V_CM2+j] = P.mb2[j]*P.mg2[j]*BNINV + P.mbe2[j];
    V[V_CB2+j] = P.bb2[j]*P.bg2[j]*BNINV + P.bbe2[j];
    float gpi = P.pig[j] * BNINV;
    for (int d = 0; d < 3; ++d) V[V_WPIX + d*64 + j] = P.Wpi[d*64+j]*gpi;
    V[V_CPI+j] = P.pib[j]*gpi + P.pibe[j];
    float gpc = P.pcg[j] * BNINV;
    for (int d = 0; d < 10; ++d) V[V_WPC + d*64 + j] = P.Wpc[d*64+j]*gpc;
    V[V_CPC+j] = P.pcb[j]*gpc + P.pcbe[j];
  }
}

// ---------------------------------------------------------------------------
// Point prep: wxyz, standardized pe/qe, normalized bf16 points, pi_enc m-part.
// ---------------------------------------------------------------------------
__global__ void prep_points(InPtrs P, char* ws) {
  int t = blockIdx.x * blockDim.x + threadIdx.x;
  if (t < 1024) {
    float lz = P.lidar_z[t];
    float* wxyz = (float*)(ws + O_WXYZ);
    for (int d = 0; d < 3; ++d) wxyz[t*3+d] = P.warped_xyz[t*3+d] * lz;
    const float* row = P.warped_points + t*64;
    float mean = 0.f, nrm = 0.f;
    for (int c = 0; c < 64; ++c) { mean += row[c]; nrm += row[c]*row[c]; }
    mean *= (1.f/64.f);
    float var = 0.f;
    for (int c = 0; c < 64; ++c) { float d = row[c]-mean; var += d*d; }
    float istd = rsqrtf(var * (1.f/63.f));
    float inv  = 1.f / fmaxf(sqrtf(nrm), 1e-12f);
    float* pe = (float*)(ws + O_PE); u16* pn = (u16*)(ws + O_PNBF);
    for (int c = 0; c < 64; ++c) {
      pe[t*64+c] = (row[c]-mean)*istd;
      pn[t*64+c] = f2bf(row[c]*inv);
    }
  } else if (t < 2048) {
    int m = t - 1024;
    const float* row = P.f2_points + m*64;
    float mean = 0.f, nrm = 0.f;
    for (int c = 0; c < 64; ++c) { mean += row[c]; nrm += row[c]*row[c]; }
    mean *= (1.f/64.f);
    float var = 0.f;
    for (int c = 0; c < 64; ++c) { float d = row[c]-mean; var += d*d; }
    float istd = rsqrtf(var * (1.f/63.f));
    float inv  = 1.f / fmaxf(sqrtf(nrm), 1e-12f);
    float* qe = (float*)(ws + O_QE); u16* qn = (u16*)(ws + O_QNBF);
    for (int c = 0; c < 64; ++c) {
      qe[m*64+c] = (row[c]-mean)*istd;
      qn[m*64+c] = f2bf(row[c]*inv);
    }
    float fx = P.f2_xyz[m*3], fy = P.f2_xyz[m*3+1], fz = P.f2_xyz[m*3+2];
    float* Q1 = (float*)(ws + O_Q1);
    for (int j = 0; j < 64; ++j) {
      float gs = P.pig[j] * BNINV;
      Q1[m*64+j] = (fx*P.Wpi[3*64+j] + fy*P.Wpi[4*64+j] + fz*P.Wpi[5*64+j]) * gs;
    }
  }
}

// ---------------------------------------------------------------------------
// cos = l2n(wp) @ l2n(f2p)^T via WMMA, 16x16 tile per wave (global operands).
// ---------------------------------------------------------------------------
__global__ __launch_bounds__(256) void cos_gemm(char* ws) {
  int wave = (blockIdx.x * blockDim.x + threadIdx.x) >> 5;
  int tn = wave >> 6, tm = wave & 63;
  const u16* pn = (const u16*)(ws + O_PNBF);
  const u16* qn = (const u16*)(ws + O_QNBF);
  f32x8 acc = {};
  acc = wmma_bf16(load_a_frag(pn + tn*16*64, 64, 0),  load_b_frag(qn, 64, tm*16, 0),  acc);
  acc = wmma_bf16(load_a_frag(pn + tn*16*64, 64, 32), load_b_frag(qn, 64, tm*16, 32), acc);
  float* cosm = (float*)(ws + O_COS);
  int lane = threadIdx.x & 31;
  int col = tm*16 + (lane & 15), rb = tn*16 + ((lane >> 4) << 3);
#pragma unroll
  for (int v = 0; v < 8; ++v) cosm[(rb + v)*1024 + col] = acc[v];
}

__global__ void rowcol_max(char* ws) {
  int t = blockIdx.x * blockDim.x + threadIdx.x;
  const float* cosm = (const float*)(ws + O_COS);
  if (t < 1024) {
    float m = -3.0e38f;
    for (int j = 0; j < 1024; ++j) m = fmaxf(m, cosm[t*1024+j]);
    ((float*)(ws + O_RMAX))[t] = m;
  } else if (t < 2048) {
    int c = t - 1024; float m = -3.0e38f;
    for (int i = 0; i < 1024; ++i) m = fmaxf(m, cosm[i*1024+c]);
    ((float*)(ws + O_CMAX))[c] = m;
  }
}

// ---------------------------------------------------------------------------
// Main fused cost-volume: one n per workgroup, flash-style online softmax.
// ---------------------------------------------------------------------------
__global__ __launch_bounds__(256) void cost_volume_main(InPtrs P, char* ws) {
  extern __shared__ char smem[];
  u16* sH   = (u16*)smem;
  u16* sWT1 = sH + LH_WT1;  u16* sWT2 = sH + LH_WT2;  u16* sWT3 = sH + LH_WT3;
  u16* sMT1 = sH + LH_MT1;  u16* sMT2 = sH + LH_MT2;  u16* sStage = sH + LH_STG;
  float* sv = (float*)(smem + L_FBASE);

  int n = blockIdx.x, tid = threadIdx.x, wave = tid >> 5, lane = tid & 31;
  const float* wxyz = (const float*)(ws + O_WXYZ);
  float wx = wxyz[n*3], wy = wxyz[n*3+1], wz = wxyz[n*3+2];

  // cooperative weight load (WT1..MT2, contiguous in ws)
  {
    const u32* src = (const u32*)(ws + O_WT1);
    u32* dst = (u32*)smem;
    for (int i = tid; i < MAIN_WCOPY_DW; i += 256) dst[i] = src[i];
  }
  const float* V = (const float*)(ws + O_VEC);
  if (tid < 128) {
    int j = tid;
    sv[FA_WA+j]=V[V_WA+j]; sv[FA_WB+j]=V[V_WB+j]; sv[FA_WC+j]=V[V_WC+j];
    sv[FA_WR+j]=V[V_WR+j]; sv[FA_WCM+j]=V[V_WCM+j]; sv[FA_C1+j]=V[V_C1+j];
    sv[FA_CM1+j]=V[V_CM1+j];
    sv[FA_X1N+j] = wx*V[V_WX0+j] + wy*V[V_WX1+j] + wz*V[V_WX2+j];
  }
  if (tid < 64) {
    int j = tid;
    sv[FA_C2+j]=V[V_C2+j]; sv[FA_C3+j]=V[V_C3+j]; sv[FA_CM2+j]=V[V_CM2+j];
    sv[FA_PEN+j] = ((const float*)(ws + O_PE))[n*64+j];
    sv[FA_P1N+j] = wx*V[V_WPIX+j] + wy*V[V_WPIX+64+j] + wz*V[V_WPIX+128+j] + V[V_CPI+j];
  }
  __syncthreads();

  float stM[4], stS[4], stA[4];
#pragma unroll
  for (int jt = 0; jt < 4; ++jt) { stM[jt] = -1e30f; stS[jt] = 0.f; stA[jt] = 0.f; }

  const float* cosrow = (const float*)(ws + O_COS) + (size_t)n * 1024;
  const float* cmax   = (const float*)(ws + O_CMAX);
  float rinv = ((const float*)(ws + O_RMAX))[n] + 1e-10f;
  const float* qe = (const float*)(ws + O_QE);
  const float* Q1 = (const float*)(ws + O_Q1);
  u16*   stg = sStage + wave * 2080;       // 16 x ST128 halves
  float* sFw = sv + FA_SF + wave * 80;     // [rm16|cm16|fx16|fy16|fz16]
  int col = lane & 15, rhalf = (lane >> 4) << 3;

  for (int chunk = 0; chunk < 8; ++chunk) {
    int m0 = (chunk * 8 + wave) * 16;
    if (chunk < 7) {  // prefetch next chunk's per-m rows into L2/L0
      int nr = m0 + 128 + (lane & 15);
      __builtin_prefetch(qe + nr*64, 0, 0);
      __builtin_prefetch(Q1 + nr*64, 0, 0);
    }
    if (lane < 16) {
      int m = m0 + lane;
      float cv = cosrow[m];
      sFw[lane]      = cv / rinv;
      sFw[16 + lane] = cv / (cmax[m] + 1e-10f);
      sFw[32 + lane] = P.f2_xyz[m*3+0];
      sFw[48 + lane] = P.f2_xyz[m*3+1];
      sFw[64 + lane] = P.f2_xyz[m*3+2];
    }
    // featdiff A tile 16x64 bf16 (stride ST64)
#pragma unroll 4
    for (int i = 0; i < 32; ++i) {
      int idx = i*32 + lane, r = idx >> 6, c = idx & 63;
      stg[r*ST64 + c] = f2bf(sv[FA_PEN + c] * qe[(m0 + r)*64 + c]);
    }
    wave_fence();
    float rm8[8], cm8[8], fx8[8], fy8[8], fz8[8];
#pragma unroll
    for (int v = 0; v < 8; ++v) {
      int r = rhalf + v;
      rm8[v]=sFw[r]; cm8[v]=sFw[16+r]; fx8[v]=sFw[32+r]; fy8[v]=sFw[48+r]; fz8[v]=sFw[64+r];
    }
    // ----- L1: 64 -> 128 (+xyz/rm/cm rank-1 epilogue) -----
    bf16x16 a0 = load_a_frag(stg, ST64, 0), a1 = load_a_frag(stg, ST64, 32);
    wave_fence();
#pragma unroll
    for (int jt = 0; jt < 8; ++jt) {
      f32x8 acc = {};
      acc = wmma_bf16(a0, load_b_frag(sWT1, ST64, jt*16, 0),  acc);
      acc = wmma_bf16(a1, load_b_frag(sWT1, ST64, jt*16, 32), acc);
      int j = jt*16 + col;
      float x1n = sv[FA_X1N+j], wa = sv[FA_WA+j], wb = sv[FA_WB+j], wc = sv[FA_WC+j];
      float wr = sv[FA_WR+j], wcm = sv[FA_WCM+j], c1 = sv[FA_C1+j];
#pragma unroll
      for (int v = 0; v < 8; ++v) {
        float pre = acc[v] + x1n + fx8[v]*wa + fy8[v]*wb + fz8[v]*wc
                  + rm8[v]*wr + cm8[v]*wcm + c1;
        stg[(rhalf+v)*ST128 + j] = f2bf(fmaxf(pre, 0.f));
      }
    }
    wave_fence();
    // ----- L2: 128 -> 64 -----
    bf16x16 af[4];
#pragma unroll
    for (int k = 0; k < 4; ++k) af[k] = load_a_frag(stg, ST128, k*32);
    wave_fence();
#pragma unroll
    for (int jt = 0; jt < 4; ++jt) {
      f32x8 acc = {};
#pragma unroll
      for (int k = 0; k < 4; ++k)
        acc = wmma_bf16(af[k], load_b_frag(sWT2, ST128, jt*16, k*32), acc);
      int j = jt*16 + col; float c2 = sv[FA_C2+j];
#pragma unroll
      for (int v = 0; v < 8; ++v)
        stg[(rhalf+v)*ST64 + j] = f2bf(fmaxf(acc[v] + c2, 0.f));
    }
    wave_fence();
    // ----- L3: 64 -> 64 (feat kept in regs + staged into concat cols 64..127)
    bf16x16 g0 = load_a_frag(stg, ST64, 0), g1 = load_a_frag(stg, ST64, 32);
    wave_fence();
    float ff[4][8];
#pragma unroll
    for (int jt = 0; jt < 4; ++jt) {
      f32x8 acc = {};
      acc = wmma_bf16(g0, load_b_frag(sWT3, ST64, jt*16, 0),  acc);
      acc = wmma_bf16(g1, load_b_frag(sWT3, ST64, jt*16, 32), acc);
      int j = jt*16 + col; float c3 = sv[FA_C3+j];
#pragma unroll
      for (int v = 0; v < 8; ++v) {
        float f = fmaxf(acc[v] + c3, 0.f);
        ff[jt][v] = f;
        stg[(rhalf+v)*ST128 + 64 + j] = f2bf(f);
      }
    }
    // enc into concat cols 0..63
#pragma unroll 4
    for (int i = 0; i < 32; ++i) {
      int idx = i*32 + lane, r = idx >> 6, c = idx & 63;
      stg[r*ST128 + c] = f2bf(fmaxf(sv[FA_P1N + c] + Q1[(m0 + r)*64 + c], 0.f));
    }
    wave_fence();
    // ----- M1: 128 -> 128 -----
    bf16x16 mf[4];
#pragma unroll
    for (int k = 0; k < 4; ++k) mf[k] = load_a_frag(stg, ST128, k*32);
    wave_fence();
#pragma unroll
    for (int jt = 0; jt < 8; ++jt) {
      f32x8 acc = {};
#pragma unroll
      for (int k = 0; k < 4; ++k)
        acc = wmma_bf16(mf[k], load_b_frag(sMT1, ST128, jt*16, k*32), acc);
      int j = jt*16 + col; float cm1 = sv[FA_CM1+j];
#pragma unroll
      for (int v = 0; v < 8; ++v)
        stg[(rhalf+v)*ST128 + j] = f2bf(fmaxf(acc[v] + cm1, 0.f));
    }
    wave_fence();
    // ----- M2: 128 -> 64, then online softmax over m -----
    bf16x16 hf[4];
#pragma unroll
    for (int k = 0; k < 4; ++k) hf[k] = load_a_frag(stg, ST128, k*32);
    wave_fence();
#pragma unroll
    for (int jt = 0; jt < 4; ++jt) {
      f32x8 acc = {};
#pragma unroll
      for (int k = 0; k < 4; ++k)
        acc = wmma_bf16(hf[k], load_b_frag(sMT2, ST128, jt*16, k*32), acc);
      int j = jt*16 + col; float cm2 = sv[FA_CM2+j];
      float hv[8];
#pragma unroll
      for (int v = 0; v < 8; ++v) hv[v] = fmaxf(acc[v] + cm2, 0.f);
      float mx = stM[jt];
#pragma unroll
      for (int v = 0; v < 8; ++v) mx = fmaxf(mx, hv[v]);
      float sc = __expf(stM[jt] - mx);
      float s = stS[jt]*sc, aa = stA[jt]*sc;
#pragma unroll
      for (int v = 0; v < 8; ++v) {
        float w = __expf(hv[v] - mx);
        s += w; aa += w * ff[jt][v];
      }
      stM[jt] = mx; stS[jt] = s; stA[jt] = aa;
    }
    wave_fence();
  }
  // merge 16 partials (8 waves x 2 lane-halves) per channel
  float* mrg = sv + FA_MRG;  // [16 partials][m64|s64|a64]
  int part = wave*2 + (lane >> 4);
#pragma unroll
  for (int jt = 0; jt < 4; ++jt) {
    int j = jt*16 + col;
    mrg[part*192 + j]       = stM[jt];
    mrg[part*192 + 64 + j]  = stS[jt];
    mrg[part*192 + 128 + j] = stA[jt];
  }
  __syncthreads();
  if (tid < 64) {
    int j = tid; float gm = -1e30f;
    for (int p = 0; p < 16; ++p) gm = fmaxf(gm, mrg[p*192 + j]);
    float s = 0.f, aa = 0.f;
    for (int p = 0; p < 16; ++p) {
      float e = __expf(mrg[p*192 + j] - gm);
      s  += mrg[p*192 + 64 + j] * e;
      aa += mrg[p*192 + 128 + j] * e;
    }
    ((float*)(ws + O_PIF))[n*64 + j] = aa / s;
  }
}

// ---------------------------------------------------------------------------
// KNN (K=16 nearest incl. self) on wxyz.
// ---------------------------------------------------------------------------
__global__ void knn_kernel(char* ws) {
  int n = blockIdx.x * blockDim.x + threadIdx.x;
  if (n >= 1024) return;
  const float* wxyz = (const float*)(ws + O_WXYZ);
  float px = wxyz[n*3], py = wxyz[n*3+1], pz = wxyz[n*3+2];
  float bd[16]; int bi[16];
  for (int k = 0; k < 16; ++k) { bd[k] = 3.0e38f; bi[k] = 0; }
  for (int m = 0; m < 1024; ++m) {
    float dx = wxyz[m*3]-px, dy = wxyz[m*3+1]-py, dz = wxyz[m*3+2]-pz;
    float d2 = dx*dx + dy*dy + dz*dz;
    if (d2 < bd[15]) {
      int pos = 15;
      while (pos > 0 && bd[pos-1] > d2) { bd[pos]=bd[pos-1]; bi[pos]=bi[pos-1]; --pos; }
      bd[pos] = d2; bi[pos] = m;
    }
  }
  int* out = (int*)(ws + O_KNN);
  for (int k = 0; k < 16; ++k) out[n*16 + k] = bi[k];
}

// ---------------------------------------------------------------------------
// Stage 2: grouped MLP (192->128->64) + softmax over K=16. One n per wave.
// ---------------------------------------------------------------------------
__global__ __launch_bounds__(256) void stage2_kernel(InPtrs P, char* ws, float* out) {
  extern __shared__ char smem[];
  u16* sH   = (u16*)smem;
  u16* sBT1 = sH + S2H_BT1;     // 128 x ST192
  u16* sBT2 = sH + S2H_BT2;     // 64 x ST128
  u16* stgAll = sH + S2H_STG;   // 8 waves * 16*ST192 halves
  float* sv = (float*)(smem + S2_FBASE);

  int tid = threadIdx.x, wave = tid >> 5, lane = tid & 31;
  int n = blockIdx.x * 8 + wave;
  {
    const u32* src = (const u32*)(ws + O_BT1);
    u32* dst = (u32*)smem;
    for (int i = tid; i < S2_WCOPY_DW; i += 256) dst[i] = src[i];
  }
  const float* V = (const float*)(ws + O_VEC);
  for (int i = tid; i < 640; i += 256) sv[i] = V[V_WPC + i];
  if (tid < 64)  sv[640 + tid] = V[V_CPC + tid];
  if (tid < 128) sv[704 + tid] = V[V_CB1 + tid];
  if (tid < 64)  sv[832 + tid] = V[V_CB2 + tid];
  __syncthreads();

  u16* stg = stgAll + wave * 3104;     // 16 x ST192 halves
  float* pw = sv + 896 + wave * 176;   // [0..159]=pc_dc(16x10) [160..175]=idx
  int* pidx = (int*)(pw + 160);
  const int* knn = (const int*)(ws + O_KNN);
  const float* wxyz = (const float*)(ws + O_WXYZ);
  const float* pif  = (const float*)(ws + O_PIF);

  if (lane < 16) {
    int r = lane, m = knn[n*16 + r];
    pidx[r] = m;
    float px = wxyz[n*3], py = wxyz[n*3+1], pz = wxyz[n*3+2];
    float gx = wxyz[m*3], gy = wxyz[m*3+1], gz = wxyz[m*3+2];
    float dx = gx-px, dy = gy-py, dz = gz-pz;
    float* d = pw + r*10;
    d[0]=px; d[1]=py; d[2]=pz; d[3]=gx; d[4]=gy; d[5]=gz;
    d[6]=dx; d[7]=dy; d[8]=dz; d[9]=sqrtf(dx*dx+dy*dy+dz*dz+1e-20f);
  }
  wave_fence();
  // concat A tile: [pc_enc(64) | warped_points[n](64) | g_feat(64)], stride ST192
  for (int i = 0; i < 96; ++i) {
    int idx = i*32 + lane, r = idx / 192, c = idx - r*192;
    float val;
    if (c < 64) {
      float s = sv[640 + c];
      const float* d = pw + r*10;
#pragma unroll
      for (int dd = 0; dd < 10; ++dd) s += d[dd] * sv[dd*64 + c];
      val = fmaxf(s, 0.f);
    } else if (c < 128) {
      val = P.warped_points[n*64 + (c - 64)];
    } else {
      val = pif[pidx[r]*64 + (c - 128)];
    }
    stg[r*ST192 + c] = f2bf(val);
  }
  wave_fence();
  int col = lane & 15, rhalf = (lane >> 4) << 3;
  bf16x16 af[6];
#pragma unroll
  for (int k = 0; k < 6; ++k) af[k] = load_a_frag(stg, ST192, k*32);
  wave_fence();
#pragma unroll
  for (int jt = 0; jt < 8; ++jt) {
    f32x8 acc = {};
#pragma unroll
    for (int k = 0; k < 6; ++k)
      acc = wmma_bf16(af[k], load_b_frag(sBT1, ST192, jt*16, k*32), acc);
    int j = jt*16 + col; float cb = sv[704 + j];
#pragma unroll
    for (int v = 0; v < 8; ++v)
      stg[(rhalf+v)*ST128 + j] = f2bf(fmaxf(acc[v] + cb, 0.f));
  }
  wave_fence();
  bf16x16 bf[4];
#pragma unroll
  for (int k = 0; k < 4; ++k) bf[k] = load_a_frag(stg, ST128, k*32);
  wave_fence();
  float* hbuf = (float*)stg;   // 16x64 f32 reuses staging
#pragma unroll
  for (int jt = 0; jt < 4; ++jt) {
    f32x8 acc = {};
#pragma unroll
    for (int k = 0; k < 4; ++k)
      acc = wmma_bf16(bf[k], load_b_frag(sBT2, ST128, jt*16, k*32), acc);
    int j = jt*16 + col; float cb2 = sv[832 + j];
#pragma unroll
    for (int v = 0; v < 8; ++v)
      hbuf[(rhalf+v)*64 + j] = fmaxf(acc[v] + cb2, 0.f);
  }
  wave_fence();
  // softmax over K=16 + weighted sum of g_feat
#pragma unroll
  for (int t = 0; t < 2; ++t) {
    int j = lane*2 + t;
    float mx = -1e30f;
    for (int r = 0; r < 16; ++r) mx = fmaxf(mx, hbuf[r*64 + j]);
    float s = 0.f, aa = 0.f;
    for (int r = 0; r < 16; ++r) {
      float w = __expf(hbuf[r*64 + j] - mx);
      s += w; aa += w * pif[pidx[r]*64 + j];
    }
    out[n*64 + j] = aa / s;
  }
}

// ---------------------------------------------------------------------------
extern "C" void kernel_launch(void* const* d_in, const int* in_sizes, int n_in,
                              void* d_out, int out_size, void* d_ws, size_t ws_size,
                              hipStream_t stream) {
  (void)in_sizes; (void)n_in; (void)out_size; (void)ws_size;
  InPtrs ip;
  __builtin_memcpy(&ip, d_in, sizeof(void*) * 41);  // 41 tensor inputs, nsample ignored (16)
  char* ws = (char*)d_ws;
  float* out = (float*)d_out;

  prep_weights<<<8, 256, 0, stream>>>(ip, ws);
  prep_points <<<8, 256, 0, stream>>>(ip, ws);
  cos_gemm    <<<512, 256, 0, stream>>>(ws);          // 4096 16x16 tiles
  rowcol_max  <<<8, 256, 0, stream>>>(ws);
  cost_volume_main<<<1024, 256, MAIN_LDS, stream>>>(ip, ws);
  knn_kernel  <<<4, 256, 0, stream>>>(ws);
  stage2_kernel<<<128, 256, S2_LDS, stream>>>(ip, ws, out);
}
